// GraphNeuralNetwork2_16277926052154
// MI455X (gfx1250) — matrix operand order repaired and checked by reading, not compiled, not run on hardware
//
#include <hip/hip_runtime.h>
#include <hip/hip_bf16.h>
#include <math.h>

typedef __attribute__((ext_vector_type(16))) _Float16 v16h;
typedef __attribute__((ext_vector_type(8)))  _Float16 v8h;
typedef __attribute__((ext_vector_type(8)))  float    v8f;

#define HDIM 128
#define LDS_LD 136  // 128 + 8 halves pad: keeps 16B alignment, rotates banks 4/row

// ---------------------------------------------------------------------------
// Pack weight W[K=128][N] (f32, row-major) into WMMA B-fragment order (f16):
//   out[nt][kc][lane][j], j=0..15 :  K = kc*32 + (lane>>4)*8 + (j&7) + (j>>3)*16
//                                    N = nt*16 + (lane&15)
// so each lane's B fragment for one 16x16x32 WMMA is 32 contiguous bytes.
// ---------------------------------------------------------------------------
__global__ void pack_w_kernel(const float* __restrict__ W,
                              _Float16* __restrict__ out, int N) {
  int idx = blockIdx.x * blockDim.x + threadIdx.x;
  int total = N * 128;
  if (idx >= total) return;
  int j    = idx & 15;
  int lane = (idx >> 4) & 31;
  int kc   = (idx >> 9) & 3;
  int nt   = idx >> 11;
  int K = kc * 32 + ((lane >> 4) * 8) + (j & 7) + ((j >> 3) * 16);
  int n = nt * 16 + (lane & 15);
  out[idx] = (_Float16)W[K * N + n];
}

__global__ void cvt_f32_f16_kernel(const float* __restrict__ in,
                                   _Float16* __restrict__ out, int total) {
  int idx = blockIdx.x * blockDim.x + threadIdx.x;
  if (idx < total) out[idx] = (_Float16)in[idx];
}

// mean over R consecutive rows (one batch) per output row; in is [nb*R][128] f16
__global__ void batch_mean_kernel(const _Float16* __restrict__ in,
                                  float* __restrict__ outF,
                                  _Float16* __restrict__ outH,
                                  int R, int total /* nb*128 */) {
  int idx = blockIdx.x * blockDim.x + threadIdx.x;
  if (idx >= total) return;
  int b = idx >> 7;
  int c = idx & 127;
  const _Float16* p = in + (size_t)b * R * HDIM + c;
  float s = 0.f;
  for (int r = 0; r < R; ++r) s += (float)p[r * HDIM];
  s /= (float)R;
  if (outF) outF[idx] = s;
  if (outH) outH[idx] = (_Float16)s;
}

// ha[row][c] = ant[row/64][c] + noise[row][c]   (rows = 8192)
__global__ void make_ha_kernel(const _Float16* __restrict__ ant,
                               const float* __restrict__ noise,
                               _Float16* __restrict__ ha, int total) {
  int idx = blockIdx.x * blockDim.x + threadIdx.x;
  if (idx >= total) return;
  int c = idx & 127;
  int b = idx >> 13;  // (idx>>7)/64
  ha[idx] = (_Float16)((float)ant[b * HDIM + c] + noise[idx]);
}

// in-place complex normalize on y[8192][16]: re=y[:8], im=y[8:]
__global__ void normalize_kernel(float* __restrict__ y, int total /* rows*8 */) {
  int idx = blockIdx.x * blockDim.x + threadIdx.x;
  if (idx >= total) return;
  int row = idx >> 3, j = idx & 7;
  float re = y[row * 16 + j];
  float im = y[row * 16 + 8 + j];
  float mag = sqrtf(re * re + im * im);
  y[row * 16 + j]     = re / mag;
  y[row * 16 + 8 + j] = im / mag;
}

// ---------------------------------------------------------------------------
// Device helpers for WMMA fragments
// ---------------------------------------------------------------------------
__device__ __forceinline__ v16h load_a_frag(const _Float16* rowPtr, int k0) {
  v8h alo = *(const v8h*)(rowPtr + k0);
  v8h ahi = *(const v8h*)(rowPtr + k0 + 16);
  v16h a;
#pragma unroll
  for (int i = 0; i < 8; ++i) { a[i] = alo[i]; a[8 + i] = ahi[i]; }
  return a;
}

__device__ __forceinline__ v16h load_b_frag(const _Float16* __restrict__ Bp,
                                            int nt, int kc, int lane) {
  const _Float16* bp = Bp + ((size_t)(nt * 4 + kc) * 32 + lane) * 16;
  v8h blo = *(const v8h*)(bp);
  v8h bhi = *(const v8h*)(bp + 8);
  v16h b;
#pragma unroll
  for (int i = 0; i < 8; ++i) { b[i] = blo[i]; b[8 + i] = bhi[i]; }
  return b;
}

// Software-pipelined K-loop: fragments for kc+1 are issued BEFORE the WMMAs of
// kc, so matrix issue overlaps the L2/LDS round-trip (partial loadcnt waits).
template <int WN>
__device__ __forceinline__ void mma_k128_pipelined(v8f (&acc)[WN],
                                                   const _Float16* Arow,
                                                   const _Float16* __restrict__ Bp,
                                                   int nt0, int g, int lane) {
  v16h aCur = load_a_frag(Arow, g * 8);
  v16h bCur[WN];
#pragma unroll
  for (int w = 0; w < WN; ++w) bCur[w] = load_b_frag(Bp, nt0 + w, 0, lane);

#pragma unroll
  for (int kc = 0; kc < 4; ++kc) {
    v16h aN;
    v16h bN[WN];
    if (kc < 3) {
      aN = load_a_frag(Arow, (kc + 1) * 32 + g * 8);
#pragma unroll
      for (int w = 0; w < WN; ++w) bN[w] = load_b_frag(Bp, nt0 + w, kc + 1, lane);
    }
#pragma unroll
    for (int w = 0; w < WN; ++w)
      acc[w] = __builtin_amdgcn_wmma_f32_16x16x32_f16(
          false, aCur, false, bCur[w], (short)0, acc[w], false, false);
    if (kc < 3) {
      aCur = aN;
#pragma unroll
      for (int w = 0; w < WN; ++w) bCur[w] = bN[w];
    }
  }
}

// ---------------------------------------------------------------------------
// Generic WMMA GEMM:  out = [relu](A[Mx128] @ W[128xN] + bias) [+ mean[batch]]
// Block = 256 threads = 8 waves, (8/NCOLW) x NCOLW grid of waves; each wave
// owns a 16 x (WN*16) output tile; K=128 as 4 x v_wmma_f32_16x16x32_f16.
// ---------------------------------------------------------------------------
template <int WN, int NCOLW>
__launch_bounds__(256)
__global__ void gemm_wmma_kernel(const _Float16* __restrict__ A,
                                 const _Float16* __restrict__ Bp,
                                 const float* __restrict__ bias,
                                 _Float16* __restrict__ outH,
                                 float* __restrict__ outF,
                                 const float* __restrict__ addMean,
                                 int rowsPerBatch, int N, int doRelu) {
  constexpr int WM = 8 / NCOLW;
  const int tid  = threadIdx.x;
  const int lane = tid & 31;
  const int wave = tid >> 5;
  const int wrow = wave / NCOLW;
  const int wcol = wave - wrow * NCOLW;
  const int mBase = blockIdx.x * (WM * 16) + wrow * 16;
  const int nBase = wcol * (WN * 16);
  const int g   = lane >> 4;
  const int l15 = lane & 15;

  v8f acc[WN] = {};
  const _Float16* Arow = A + (size_t)(mBase + l15) * HDIM;
  mma_k128_pipelined<WN>(acc, Arow, Bp, nBase >> 4, g, lane);

#pragma unroll
  for (int w = 0; w < WN; ++w) {
    const int n = nBase + w * 16 + l15;
    const float bval = bias[n];
#pragma unroll
    for (int r = 0; r < 8; ++r) {
      const int m = mBase + g * 8 + r;
      float x = acc[w][r] + bval;
      if (doRelu) x = fmaxf(x, 0.f);
      if (addMean) x += addMean[(size_t)(m / rowsPerBatch) * HDIM + n];
      if (outH) outH[(size_t)m * N + n] = (_Float16)x;
      if (outF) outF[(size_t)m * N + n] = x;
    }
  }
}

// ---------------------------------------------------------------------------
// Fused 2-layer MLP (all conv MLPs have NL=2, K=N=128):
//   out = relu(relu(A@W0+b0)@W1+b1) [+ mean[batch] after final relu]
// Block tile 64(M) x 128(N); layer-0 output kept in LDS (padded rows), so the
// intermediate never touches global memory. 32 WMMAs per wave per kernel.
// ---------------------------------------------------------------------------
__launch_bounds__(256)
__global__ void fused_mlp2_kernel(const _Float16* __restrict__ A,
                                  const _Float16* __restrict__ Bp0,
                                  const float* __restrict__ bias0,
                                  const _Float16* __restrict__ Bp1,
                                  const float* __restrict__ bias1,
                                  _Float16* __restrict__ outH,
                                  const float* __restrict__ addMean,
                                  int rowsPerBatch) {
  __shared__ _Float16 smem[64 * LDS_LD];

  const int tid  = threadIdx.x;
  const int lane = tid & 31;
  const int wave = tid >> 5;
  const int wrow = wave >> 1;   // 4 wave-rows x 2 wave-cols
  const int wcol = wave & 1;
  const int mBase = blockIdx.x * 64 + wrow * 16;
  const int nBase = wcol * 64;
  const int g   = lane >> 4;
  const int l15 = lane & 15;

  // ---- layer 0: global A -> WMMA -> relu+bias -> LDS ----
  {
    v8f acc[4] = {};
    const _Float16* Arow = A + (size_t)(mBase + l15) * HDIM;
    mma_k128_pipelined<4>(acc, Arow, Bp0, nBase >> 4, g, lane);
#pragma unroll
    for (int w = 0; w < 4; ++w) {
      const int n = nBase + w * 16 + l15;
      const float bv = bias0[n];
#pragma unroll
      for (int r = 0; r < 8; ++r) {
        const int ml = wrow * 16 + g * 8 + r;  // row within block tile
        smem[ml * LDS_LD + n] = (_Float16)fmaxf(acc[w][r] + bv, 0.f);
      }
    }
  }
  __syncthreads();

  // ---- layer 1: LDS A -> WMMA -> relu+bias (+mean) -> global ----
  {
    v8f acc[4] = {};
    const _Float16* Lrow = smem + (size_t)(wrow * 16 + l15) * LDS_LD;
    mma_k128_pipelined<4>(acc, Lrow, Bp1, nBase >> 4, g, lane);
#pragma unroll
    for (int w = 0; w < 4; ++w) {
      const int n = nBase + w * 16 + l15;
      const float bv = bias1[n];
#pragma unroll
      for (int r = 0; r < 8; ++r) {
        const int m = mBase + g * 8 + r;
        float x = fmaxf(acc[w][r] + bv, 0.f);
        if (addMean) x += addMean[(size_t)(m / rowsPerBatch) * HDIM + n];
        outH[(size_t)m * HDIM + n] = (_Float16)x;
      }
    }
  }
}

// ---------------------------------------------------------------------------
extern "C" void kernel_launch(void* const* d_in, const int* in_sizes, int n_in,
                              void* d_out, int out_size, void* d_ws, size_t ws_size,
                              hipStream_t stream) {
  (void)in_sizes; (void)n_in; (void)out_size; (void)ws_size;

  const float* user_feat = (const float*)d_in[0];   // [4096][128]
  const float* ant_noise = (const float*)d_in[1];   // [8192][128]
  // d_in[2..5]: edge lists unused — graph is complete-bipartite per batch,
  // so segment sums reduce to dense per-batch means.
  const float* W_ue = (const float*)d_in[6];
  const float* b_ue = (const float*)d_in[7];
  const float* W_t  = (const float*)d_in[8];
  const float* b_t  = (const float*)d_in[9];
  const float* convW[6]; const float* convB[6];
  for (int i = 0; i < 6; ++i) {  // ca_aggr, ca_self, ca_comb, cu_aggr, cu_self, cu_comb
    convW[i] = (const float*)d_in[10 + 2 * i];
    convB[i] = (const float*)d_in[11 + 2 * i];
  }
  const float* W_norm = (const float*)d_in[22];
  const float* b_norm = (const float*)d_in[23];

  // -------- workspace layout --------
  char* ws = (char*)d_ws;
  size_t off = 0;
  auto alloc = [&](size_t bytes) -> char* {
    char* p = ws + off;
    off += (bytes + 255) & ~(size_t)255;
    return p;
  };

  _Float16* pw[14];  // 0:W_ue 1:W_t 2+i*2+l: conv name i, layer l
  for (int i = 0; i < 14; ++i) pw[i] = (_Float16*)alloc(16384 * sizeof(_Float16));
  _Float16* pw_norm = (_Float16*)alloc(2048 * sizeof(_Float16));

  _Float16* uf16  = (_Float16*)alloc((size_t)4096 * 128 * 2);
  _Float16* hu[2] = {(_Float16*)alloc((size_t)4096 * 128 * 2),
                     (_Float16*)alloc((size_t)4096 * 128 * 2)};
  _Float16* ha[2] = {(_Float16*)alloc((size_t)8192 * 128 * 2),
                     (_Float16*)alloc((size_t)8192 * 128 * 2)};
  _Float16* tB    = (_Float16*)alloc((size_t)8192 * 128 * 2);
  float*    meanF = (float*)alloc((size_t)128 * 128 * 4);
  _Float16* meanH = (_Float16*)alloc((size_t)128 * 128 * 2);
  _Float16* ant16 = (_Float16*)alloc((size_t)128 * 128 * 2);

  // -------- pack all weights into WMMA B-fragment order (f16) --------
  auto packW = [&](const float* W, _Float16* dst, int N) {
    int total = N * 128;
    pack_w_kernel<<<(total + 255) / 256, 256, 0, stream>>>(W, dst, N);
  };
  packW(W_ue, pw[0], 128);
  packW(W_t,  pw[1], 128);
  for (int i = 0; i < 6; ++i)
    for (int l = 0; l < 2; ++l) packW(convW[i] + (size_t)l * 16384, pw[2 + i * 2 + l], 128);
  packW(W_norm, pw_norm, 16);

  auto gemm128 = [&](const _Float16* Ain, const _Float16* Bp, const float* bias,
                     _Float16* outH, int M) {
    gemm_wmma_kernel<4, 2><<<dim3(M / 64), dim3(256), 0, stream>>>(
        Ain, Bp, bias, outH, (float*)nullptr, (const float*)nullptr, 1, 128, 1);
  };
  auto mlp2 = [&](const _Float16* Ain, int wi /* conv name index */,
                  _Float16* outH, int M, const float* addMean, int rpb) {
    fused_mlp2_kernel<<<dim3(M / 64), dim3(256), 0, stream>>>(
        Ain, pw[2 + wi * 2], convB[wi], pw[2 + wi * 2 + 1], convB[wi] + 128,
        outH, addMean, rpb);
  };
  auto bmean = [&](const _Float16* in, int R, float* oF, _Float16* oH) {
    int total = 128 * 128;
    batch_mean_kernel<<<(total + 255) / 256, 256, 0, stream>>>(in, oF, oH, R, total);
  };

  // -------- prologue --------
  cvt_f32_f16_kernel<<<(4096 * 128 + 255) / 256, 256, 0, stream>>>(
      user_feat, uf16, 4096 * 128);
  gemm128(uf16, pw[0], b_ue, hu[0], 4096);           // hu = relu(uf @ W_ue + b)
  bmean(hu[0], 32, meanF, meanH);                    // mean over users per batch
  gemm128(meanH, pw[1], b_t, ant16, 128);            // ant = relu(mean @ W_t + b)
  make_ha_kernel<<<(8192 * 128 + 255) / 256, 256, 0, stream>>>(
      ant16, ant_noise, ha[0], 8192 * 128);          // ha = repeat(ant,64) + noise

  // -------- 2 message-passing iterations --------
  int cur = 0;
  for (int it = 0; it < 2; ++it) {
    int nxt = cur ^ 1;
    // antenna update (ca_*)
    mlp2(hu[cur], 0, tB, 4096, nullptr, 1);          // msg  = mlp(hu, ca_aggr)
    bmean(tB, 32, meanF, nullptr);                   // mean over 32 users
    mlp2(ha[cur], 1, tB, 8192, meanF, 64);           // self = mlp(ha, ca_self) + mean
    mlp2(tB, 2, ha[nxt], 8192, nullptr, 1);          // na   = mlp(self, ca_comb)
    // user update (cu_*), uses OLD ha/hu
    mlp2(ha[cur], 3, tB, 8192, nullptr, 1);          // msg  = mlp(ha, cu_aggr)
    bmean(tB, 64, meanF, nullptr);                   // mean over 64 antennas
    mlp2(hu[cur], 4, tB, 4096, meanF, 32);           // self = mlp(hu, cu_self) + mean
    mlp2(tB, 5, hu[nxt], 4096, nullptr, 1);          // nu   = mlp(self, cu_comb)
    cur = nxt;
  }

  // -------- epilogue: y = ha @ W_norm + b_norm ; complex normalize --------
  gemm_wmma_kernel<1, 1><<<dim3(8192 / 128), dim3(256), 0, stream>>>(
      ha[cur], pw_norm, b_norm, (_Float16*)nullptr, (float*)d_out,
      (const float*)nullptr, 1, 16, 0);
  normalize_kernel<<<(8192 * 8 + 255) / 256, 256, 0, stream>>>((float*)d_out, 8192 * 8);
}
